// N3AggregationBase_34943853920738
// MI455X (gfx1250) — matrix operand order repaired and checked by reading, not compiled, not run on hardware
//
#include <hip/hip_runtime.h>
#include <hip/hip_bf16.h>

typedef __attribute__((ext_vector_type(16))) _Float16 v16h;
typedef __attribute__((ext_vector_type(8)))  _Float16 v8h;
typedef __attribute__((ext_vector_type(8)))  float    v8f;

#define DIST_SCALE 10.0f

constexpr int C = 4, H = 256, W = 256, PS = 7, STRIDE = 2, E = 32, O = 32;
constexpr int NSIDE = (H - PS) / STRIDE + 1;   // 125
constexpr int NQ    = NSIDE * NSIDE;           // 15625
constexpr int D     = C * PS * PS;             // 196
constexpr int DPAD  = 208;                     // 13 chunks of 16
constexpr int WAVES = 4;                       // queries per block

__global__ void zero_out_kernel(float* out, int nelem) {
  int i = blockIdx.x * blockDim.x + threadIdx.x;
  if (i < nelem) out[i] = 0.0f;
}

__global__ __launch_bounds__(WAVES * 32)
void n3agg_kernel(const float* __restrict__ x,
                  const float* __restrict__ xe,
                  const float* __restrict__ ye,
                  const int*   __restrict__ inds,
                  float* __restrict__ out) {
  // Per-wave scratch: Pt[d][o] (f16, transposed patches) + z accumulator.
  __shared__ alignas(64) _Float16 sP[WAVES][DPAD * 32];   // 13312 B / wave
  __shared__ float                sz[WAVES][DPAD];        //   832 B / wave

  const int lane = threadIdx.x & 31;
  const int wid  = threadIdx.x >> 5;
  const int q    = blockIdx.x * WAVES + wid;
  if (q >= NQ) return;                  // whole-wave uniform -> EXEC stays full

  _Float16* Pt = sP[wid];
  float*    zb = sz[wid];

  // ---- 1. distance + softmax: lane o handles neighbor o (O == wave32) ----
  const int ind = inds[(size_t)q * O + lane];
  float d2 = 0.0f;
  const float4* yv = (const float4*)(ye + (size_t)q   * E);
  const float4* xv = (const float4*)(xe + (size_t)ind * E);
  #pragma unroll
  for (int t = 0; t < E / 4; ++t) {
    float4 a = yv[t], b = xv[t];
    float dx = a.x - b.x, dy = a.y - b.y, dz = a.z - b.z, dw = a.w - b.w;
    d2 += dx * dx + dy * dy + dz * dz + dw * dw;
  }
  float s = -DIST_SCALE * d2;
  float mx = s;
  #pragma unroll
  for (int off = 16; off > 0; off >>= 1) mx = fmaxf(mx, __shfl_xor(mx, off, 32));
  float ex = __expf(s - mx);
  float sum = ex;
  #pragma unroll
  for (int off = 16; off > 0; off >>= 1) sum += __shfl_xor(sum, off, 32);
  const float w = ex / sum;            // softmax weight for this lane's neighbor

  // ---- 2. gather 32 neighbor patches -> LDS, transposed + f16 ----
  {
    const int pr = ind / NSIDE, pc = ind % NSIDE;
    const float* base0 = x + (size_t)(pr * STRIDE) * W + (pc * STRIDE);
    int d = 0;
    #pragma unroll
    for (int ch = 0; ch < C; ++ch) {
      const float* base = base0 + (size_t)ch * H * W;
      #pragma unroll
      for (int i = 0; i < PS; ++i) {
        const float* row = base + i * W;
        #pragma unroll
        for (int j = 0; j < PS; ++j) {
          Pt[d * 32 + lane] = (_Float16)row[j];  // wave writes 64B contiguous
          ++d;
        }
      }
    }
    for (; d < DPAD; ++d) Pt[d * 32 + lane] = (_Float16)0.0f;
  }
  asm volatile("s_wait_dscnt 0" ::: "memory");   // LDS writes visible to reads

  // ---- 3. z = w^T * P via 13x v_wmma_f32_16x16x32_f16 ----
  // B layout (table order): element t of lane half h holds w of neighbor
  // K = 8h+t (t<8) or 16+8h+(t-8) (t>=8), broadcast over all N-columns.
  v16h bmat;
  {
    #pragma unroll
    for (int t = 0; t < 16; ++t) bmat[t] = (_Float16)__shfl(w, (t < 8 ? 0 : 8) + (lane >> 4) * 8 + t, 32);
  }
  // A layout (16-bit 16x32): lanes 0-15 row m, K=0..7 & 16..23;
  //                          lanes 16-31 row m, K=8..15 & 24..31.
  const int mrow = lane & 15;
  const int ks0  = (lane < 16) ? 0 : 8;

  #pragma unroll
  for (int c = 0; c < DPAD / 16; ++c) {
    const int d0 = c * 16;
    const v8h* rowp = (const v8h*)&Pt[(d0 + mrow) * 32];   // 64B-aligned row
    v8h lo = rowp[ks0 >> 3];             // halves ks0   .. ks0+7
    v8h hi = rowp[(ks0 + 16) >> 3];      // halves ks0+16.. ks0+23
    v16h a;
    #pragma unroll
    for (int t = 0; t < 8; ++t) { a[t] = lo[t]; a[8 + t] = hi[t]; }
    v8f cz = {};
    v8f acc = __builtin_amdgcn_wmma_f32_16x16x32_f16(
        /*neg_a=*/false, a, /*neg_b=*/false, bmat,
        /*c_mod=*/(short)0, cz, /*reuse_a=*/false, /*reuse_b=*/false);
    // C layout: lanes 0-15 hold z[d0+v] in VGPR v; lanes 16-31 hold z[d0+8+v].
    if (lane == 0) {
      #pragma unroll
      for (int v = 0; v < 8; ++v) zb[d0 + v] = acc[v];
    } else if (lane == 16) {
      #pragma unroll
      for (int v = 0; v < 8; ++v) zb[d0 + 8 + v] = acc[v];
    }
  }
  asm volatile("s_wait_dscnt 0" ::: "memory");

  // ---- 4. fold: overlap-add z back at this query's patch position ----
  const int qr = q / NSIDE, qc = q % NSIDE;
  const int r0 = qr * STRIDE, c0 = qc * STRIDE;
  for (int d = lane; d < D; d += 32) {
    const int ch  = d / (PS * PS);
    const int rem = d - ch * PS * PS;
    const int i = rem / PS, j = rem - i * PS;
    float* dst = out + (size_t)ch * H * W + (size_t)(r0 + i) * W + (c0 + j);
    __hip_atomic_fetch_add(dst, zb[d], __ATOMIC_RELAXED,
                           __HIP_MEMORY_SCOPE_AGENT);
  }
}

extern "C" void kernel_launch(void* const* d_in, const int* in_sizes, int n_in,
                              void* d_out, int out_size, void* d_ws, size_t ws_size,
                              hipStream_t stream) {
  const float* x    = (const float*)d_in[0];
  const float* xe   = (const float*)d_in[1];
  const float* ye   = (const float*)d_in[2];
  const int*   inds = (const int*)  d_in[3];
  // d_in[4]=ps, d_in[5]=stride are compile-time constants here (7, 2).
  float* out = (float*)d_out;

  const int nelem = C * H * W;                       // 262144
  zero_out_kernel<<<(nelem + 255) / 256, 256, 0, stream>>>(out, nelem);

  const int blocks = (NQ + WAVES - 1) / WAVES;       // 3907
  n3agg_kernel<<<blocks, WAVES * 32, 0, stream>>>(x, xe, ye, inds, out);
}
